// Attention_77575699300898
// MI455X (gfx1250) — matrix operand order repaired and checked by reading, not compiled, not run on hardware
//
#include <hip/hip_runtime.h>
#include <cstdint>
#include <cstddef>

// ---------------------------------------------------------------------------
// Types / helpers
// ---------------------------------------------------------------------------
typedef __attribute__((ext_vector_type(16))) __bf16      v16bf;
typedef __attribute__((ext_vector_type(8)))  float       v8f;
typedef __attribute__((ext_vector_type(8)))  unsigned    v8u;

__device__ __forceinline__ __bf16 f2bf(float f) {
  unsigned u = __builtin_bit_cast(unsigned, f);
  unsigned r = u + 0x7FFFu + ((u >> 16) & 1u);          // round-to-nearest-even
  unsigned short s = (unsigned short)(r >> 16);
  return __builtin_bit_cast(__bf16, s);
}
__device__ __forceinline__ float bf2f(__bf16 b) {
  unsigned u = ((unsigned)__builtin_bit_cast(unsigned short, b)) << 16;
  return __builtin_bit_cast(float, u);
}
__device__ __forceinline__ __bf16 bf_zero() {
  return __builtin_bit_cast(__bf16, (unsigned short)0);
}
__device__ __forceinline__ v8f wmma_bf16(v16bf a, v16bf b, v8f c) {
  // D = A(16x32 bf16) * B(32x16 bf16) + C(16x16 f32)
  return __builtin_amdgcn_wmma_f32_16x16x32_bf16(false, a, false, b,
                                                 (short)0, c, false, false);
}

// --- CDNA5 async global->LDS copy (ASYNCcnt-tracked) via inline asm --------
#if defined(__gfx1250__)
#define HAVE_ASYNC_LDS 1
#else
#define HAVE_ASYNC_LDS 0
#endif

#if HAVE_ASYNC_LDS && __has_builtin(__builtin_amdgcn_s_wait_asynccnt)
#define WAIT_ASYNC() __builtin_amdgcn_s_wait_asynccnt(0)
#elif HAVE_ASYNC_LDS
#define WAIT_ASYNC() asm volatile("s_wait_asynccnt 0x0" ::: "memory")
#else
#define WAIT_ASYNC() ((void)0)
#endif

__device__ __forceinline__ void async_copy16(void* lds_dst, const void* gsrc) {
#if HAVE_ASYNC_LDS
  // VFLAT async load: VDST = LDS byte address, VADDR = 64-bit global address.
  asm volatile("global_load_async_to_lds_b128 %0, %1, off"
               :
               : "v"((unsigned)(uintptr_t)lds_dst),
                 "v"((unsigned long long)(uintptr_t)gsrc)
               : "memory");
#else
  *(uint4*)lds_dst = *(const uint4*)gsrc;
#endif
}

// Problem constants
#define S_LEN 2048
#define D_MODEL 2880
#define N_HEADS 64
#define KV_HEADS 8
#define HEAD_DIM 64
#define QDIM (N_HEADS * HEAD_DIM)   // 4096
#define KVDIM (KV_HEADS * HEAD_DIM) // 512
#define WINDOW 128

// ---------------------------------------------------------------------------
// fp32 -> bf16 conversion
// ---------------------------------------------------------------------------
__global__ void cvt_f32_bf16(const float* __restrict__ in,
                             __bf16* __restrict__ out, int n) {
  int i = blockIdx.x * blockDim.x + threadIdx.x;
  int stride = gridDim.x * blockDim.x;
  for (; i < n; i += stride) out[i] = f2bf(in[i]);
}

// ---------------------------------------------------------------------------
// GEMM:  C[M,N] = A[M,K](bf16,row) * W[N,K]^T(bf16,row) + bias[N]
// Block tile 128x64, 8 waves (4x2), each wave 32x32 (2x2 WMMA tiles).
// Double-buffered LDS slabs filled with async global->LDS copies.
// ---------------------------------------------------------------------------
template <bool OUT_BF16>
__global__ __launch_bounds__(256) void gemm_xwT(
    const __bf16* __restrict__ A, const __bf16* __restrict__ W,
    const float* __restrict__ bias, void* __restrict__ Cout,
    int M, int N, int K) {
  __shared__ __align__(32) __bf16 Al[2][128][32];
  __shared__ __align__(32) __bf16 Bl[2][64][32];

  const int tid  = threadIdx.x;
  const int lane = tid & 31;
  const int wid  = tid >> 5;
  const int wm   = wid & 3;   // wave row quadrant (0..3)
  const int wn   = wid >> 2;  // wave col half    (0..1)
  const int m0   = blockIdx.y * 128;
  const int n0   = blockIdx.x * 64;
  const int l15  = lane & 15;
  const int lhi  = lane >> 4; // 0/1

  const int arow = tid >> 1, achk = tid & 1;  // A staging: 128 rows x 2 chunks
  const int brow = tid >> 2, bchk = tid & 3;  // B staging: 64 rows x 4 chunks

  auto stage = [&](int buf, int k0) {
    const __bf16* as = A + (size_t)(m0 + arow) * K + k0 + achk * 16;
    async_copy16(&Al[buf][arow][achk * 16], as);
    async_copy16(&Al[buf][arow][achk * 16 + 8], as + 8);
    const __bf16* ws = W + (size_t)(n0 + brow) * K + k0 + bchk * 8;
    async_copy16(&Bl[buf][brow][bchk * 8], ws);
  };

  v8f acc[2][2] = {};
  const int nk = K >> 5;

  stage(0, 0);
  WAIT_ASYNC();
  __syncthreads();

  for (int ik = 0; ik < nk; ++ik) {
    const int cur = ik & 1;
    if (ik + 1 < nk) stage(cur ^ 1, (ik + 1) << 5);  // overlap next slab
    if (ik + 2 < nk) {                               // L2 prefetch 2 ahead
      __builtin_prefetch(A + (size_t)(m0 + arow) * K + ((ik + 2) << 5) + achk * 16, 0, 1);
      __builtin_prefetch(W + (size_t)(n0 + brow) * K + ((ik + 2) << 5) + bchk * 8, 0, 1);
    }

    // Build fragments (this code's contiguous k = 16*(lane/16) + s order in both A and B).
    v8u au[2];
    v16bf bfr[2];
#pragma unroll
    for (int mt = 0; mt < 2; ++mt) {
      const int row = wm * 32 + mt * 16 + l15;
#pragma unroll
      for (int v = 0; v < 8; ++v) {
        const int kk = 2 * v + lhi * 16;
        au[mt][v] = *(const unsigned*)(&Al[cur][row][kk]);
      }
    }
#pragma unroll
    for (int nt = 0; nt < 2; ++nt) {
      const int row = wn * 32 + nt * 16 + l15;
      bfr[nt] = *(const v16bf*)(&Bl[cur][row][lhi * 16]);
    }
#pragma unroll
    for (int mt = 0; mt < 2; ++mt) {
      const v16bf av = __builtin_bit_cast(v16bf, au[mt]);
#pragma unroll
      for (int nt = 0; nt < 2; ++nt)
        acc[mt][nt] = wmma_bf16(av, bfr[nt], acc[mt][nt]);
    }

    WAIT_ASYNC();      // our async writes to the next buffer have landed
    __syncthreads();   // everyone done reading cur / writing next
  }

  // Epilogue: C layout => vgpr r, lane<16: M=r ; lane>=16: M=r+8 ; N=lane&15
#pragma unroll
  for (int mt = 0; mt < 2; ++mt) {
#pragma unroll
    for (int nt = 0; nt < 2; ++nt) {
      const int n = n0 + wn * 32 + nt * 16 + l15;
      const float b = bias[n];
#pragma unroll
      for (int r = 0; r < 8; ++r) {
        const int m = m0 + wm * 32 + mt * 16 + r + lhi * 8;
        const float val = acc[mt][nt][r] + b;
        if (OUT_BF16)
          ((__bf16*)Cout)[(size_t)m * N + n] = f2bf(val);
        else
          ((float*)Cout)[(size_t)m * N + n] = val;
      }
    }
  }
}

// ---------------------------------------------------------------------------
// RoPE (in place on bf16 [S, nh, 64]); rope_cache: [S, 2, 64] f32
// ---------------------------------------------------------------------------
__global__ void rope_kernel(__bf16* __restrict__ t,
                            const float* __restrict__ rc, int nh, int total) {
  int idx = blockIdx.x * blockDim.x + threadIdx.x;
  if (idx >= total) return;
  const int d = idx & 31;
  const int h = (idx >> 5) % nh;
  const int i = idx / (32 * nh);
  const size_t o = (size_t)i * nh * 64 + h * 64 + d;
  const float c = rc[i * 128 + d];       // cos (same for d and d+32)
  const float s = rc[i * 128 + 64 + d];  // sin
  const float x1 = bf2f(t[o]);
  const float x2 = bf2f(t[o + 32]);
  t[o]      = f2bf(x1 * c - x2 * s);
  t[o + 32] = f2bf(x2 * c + x1 * s);
}

// ---------------------------------------------------------------------------
// Sliding-window attention. grid = (S/16, KV_HEADS), 256 threads (8 waves).
// Wave w handles head h = kvh*8 + w for query tile qt = blockIdx.x.
// ---------------------------------------------------------------------------
__global__ __launch_bounds__(256) void attn_kernel(
    const __bf16* __restrict__ qb,  // [S, 4096]
    const __bf16* __restrict__ kb,  // [S, 512]
    const __bf16* __restrict__ vb,  // [S, 512]
    __bf16* __restrict__ ao) {      // [S, 4096]
  __shared__ __align__(32) __bf16 Ksh[160][64];   // keys x dims
  __shared__ __align__(32) __bf16 Vt[64][160];    // dims x keys (transposed)
  __shared__ __align__(32) __bf16 Pp[8][16][32];  // per-wave P pair tile

  const int qt   = blockIdx.x;
  const int kvh  = blockIdx.y;
  const int tid  = threadIdx.x;
  const int lane = tid & 31;
  const int wid  = tid >> 5;
  const int h    = kvh * 8 + wid;
  const int base = qt - 9;  // 10 key tiles: base .. qt (pair-aligned)
  const int l15  = lane & 15;
  const int lhi  = lane >> 4;

  // K strip via async global->LDS (16B chunks); zero-fill negative rows.
  for (int idx = tid; idx < 1280; idx += 256) {
    const int row = idx >> 3;          // key 0..159
    const int d0  = (idx & 7) * 8;     // dim chunk
    const int j   = base * 16 + row;
    if (j >= 0) {
      async_copy16(&Ksh[row][d0], kb + (size_t)j * KVDIM + kvh * 64 + d0);
    } else {
      uint4 z{};
      *(uint4*)(&Ksh[row][d0]) = z;
    }
  }
  // Transposed V strip (manual transpose).
  for (int idx = tid; idx < 160 * 64; idx += 256) {
    const int row = idx >> 6;
    const int d   = idx & 63;
    const int j   = base * 16 + row;
    Vt[d][row] = (j >= 0) ? vb[(size_t)j * KVDIM + kvh * 64 + d] : bf_zero();
  }
  WAIT_ASYNC();
  __syncthreads();

  // Q A-fragments (two 16x32 K-slices of HEAD_DIM=64) straight from global.
  v8u qu[2];
  const __bf16* qrow = qb + (size_t)(qt * 16 + l15) * QDIM + h * 64;
#pragma unroll
  for (int half = 0; half < 2; ++half)
#pragma unroll
    for (int v = 0; v < 8; ++v) {
      const int kk = 2 * v + lhi * 16 + half * 32;
      qu[half][v] = *(const unsigned*)(qrow + kk);
    }
  const v16bf qf0 = __builtin_bit_cast(v16bf, qu[0]);
  const v16bf qf1 = __builtin_bit_cast(v16bf, qu[1]);

  // Pass 1: scores for all key tiles, masked; track per-row max.
  v8f sc[10];
  float rmax[8];
#pragma unroll
  for (int r = 0; r < 8; ++r) rmax[r] = -1e30f;

#pragma unroll
  for (int t = 0; t < 10; ++t) {
    const int kt = base + t;
    if (t >= 1 && kt >= 0) {
      const v16bf kf0 = *(const v16bf*)(&Ksh[t * 16 + l15][lhi * 16]);
      const v16bf kf1 = *(const v16bf*)(&Ksh[t * 16 + l15][32 + lhi * 16]);
      v8f s = {};
      s = wmma_bf16(qf0, kf0, s);
      s = wmma_bf16(qf1, kf1, s);
      const int j = kt * 16 + l15;
#pragma unroll
      for (int r = 0; r < 8; ++r) {
        const int m = qt * 16 + r + lhi * 8;
        float val = s[r] * 0.125f;  // 1/sqrt(64)
        const bool ok = (j <= m) && ((m - j) < WINDOW);
        val = ok ? val : -1e30f;
        sc[t][r] = val;
        rmax[r] = fmaxf(rmax[r], val);
      }
    } else {
#pragma unroll
      for (int r = 0; r < 8; ++r) sc[t][r] = -1e30f;
    }
  }

  // Row max across the 16 lanes holding each row (xor masks stay in-half).
#pragma unroll
  for (int off = 1; off <= 8; off <<= 1)
#pragma unroll
    for (int r = 0; r < 8; ++r)
      rmax[r] = fmaxf(rmax[r], __shfl_xor(rmax[r], off, 32));

  // Pass 2: exp, P->LDS relayout, P*V accumulation.
  float rsum[8] = {0.f, 0.f, 0.f, 0.f, 0.f, 0.f, 0.f, 0.f};
  v8f acc[4] = {};

#pragma unroll
  for (int p = 0; p < 5; ++p) {
#pragma unroll
    for (int tt = 0; tt < 2; ++tt) {
      const int t = 2 * p + tt;
#pragma unroll
      for (int r = 0; r < 8; ++r) {
        const float e = __expf(sc[t][r] - rmax[r]);
        rsum[r] += e;
        Pp[wid][r + lhi * 8][tt * 16 + l15] = f2bf(e);  // D-layout -> tile
      }
    }
    // A-fragment of P (wave-private LDS; DS ops in-order within wave).
    v8u pu;
#pragma unroll
    for (int v = 0; v < 8; ++v) {
      const int kk = 2 * v + lhi * 16;
      pu[v] = *(const unsigned*)(&Pp[wid][l15][kk]);
    }
    const v16bf pf = __builtin_bit_cast(v16bf, pu);
#pragma unroll
    for (int dt = 0; dt < 4; ++dt) {
      const v16bf vf = *(const v16bf*)(&Vt[dt * 16 + l15][p * 32 + lhi * 16]);
      acc[dt] = wmma_bf16(pf, vf, acc[dt]);
    }
  }

  // Row sums and normalization.
#pragma unroll
  for (int off = 1; off <= 8; off <<= 1)
#pragma unroll
    for (int r = 0; r < 8; ++r) rsum[r] += __shfl_xor(rsum[r], off, 32);
  float inv[8];
#pragma unroll
  for (int r = 0; r < 8; ++r) inv[r] = 1.0f / rsum[r];

#pragma unroll
  for (int dt = 0; dt < 4; ++dt)
#pragma unroll
    for (int r = 0; r < 8; ++r) {
      const size_t m = (size_t)(qt * 16 + r + lhi * 8);
      ao[m * QDIM + h * 64 + dt * 16 + l15] = f2bf(acc[dt][r] * inv[r]);
    }
}

// ---------------------------------------------------------------------------
// Host launch
// ---------------------------------------------------------------------------
extern "C" void kernel_launch(void* const* d_in, const int* in_sizes, int n_in,
                              void* d_out, int out_size, void* d_ws,
                              size_t ws_size, hipStream_t stream) {
  (void)in_sizes; (void)n_in; (void)out_size; (void)ws_size;
  const float* x_f   = (const float*)d_in[0];
  const float* rope  = (const float*)d_in[1];
  const float* wq_w  = (const float*)d_in[2];
  const float* wq_b  = (const float*)d_in[3];
  const float* wk_w  = (const float*)d_in[4];
  const float* wk_b  = (const float*)d_in[5];
  const float* wv_w  = (const float*)d_in[6];
  const float* wv_b  = (const float*)d_in[7];
  const float* wo_w  = (const float*)d_in[8];
  const float* wo_b  = (const float*)d_in[9];
  float* out = (float*)d_out;

  // Workspace layout (bf16 elements).
  const size_t SZ_X  = (size_t)S_LEN * D_MODEL;
  const size_t SZ_WQ = (size_t)QDIM * D_MODEL;
  const size_t SZ_WK = (size_t)KVDIM * D_MODEL;
  const size_t SZ_WO = (size_t)D_MODEL * QDIM;
  const size_t SZ_Q  = (size_t)S_LEN * QDIM;
  const size_t SZ_KV = (size_t)S_LEN * KVDIM;

  __bf16* xb  = (__bf16*)d_ws;
  __bf16* wqb = xb + SZ_X;
  __bf16* wkb = wqb + SZ_WQ;
  __bf16* wvb = wkb + SZ_WK;
  __bf16* wob = wvb + SZ_WK;
  __bf16* qb  = wob + SZ_WO;
  __bf16* kbf = qb + SZ_Q;
  __bf16* vbf = kbf + SZ_KV;
  __bf16* aob = vbf + SZ_KV;

  auto blocks = [](size_t n) { return (unsigned)((n + 255) / 256); };

  // 1) fp32 -> bf16 conversions
  cvt_f32_bf16<<<blocks(SZ_X), 256, 0, stream>>>(x_f, xb, (int)SZ_X);
  cvt_f32_bf16<<<blocks(SZ_WQ), 256, 0, stream>>>(wq_w, wqb, (int)SZ_WQ);
  cvt_f32_bf16<<<blocks(SZ_WK), 256, 0, stream>>>(wk_w, wkb, (int)SZ_WK);
  cvt_f32_bf16<<<blocks(SZ_WK), 256, 0, stream>>>(wv_w, wvb, (int)SZ_WK);
  cvt_f32_bf16<<<blocks(SZ_WO), 256, 0, stream>>>(wo_w, wob, (int)SZ_WO);

  // 2) QKV projections (WMMA bf16, bias fused, bf16 out)
  gemm_xwT<true><<<dim3(QDIM / 64, S_LEN / 128), 256, 0, stream>>>(
      xb, wqb, wq_b, qb, S_LEN, QDIM, D_MODEL);
  gemm_xwT<true><<<dim3(KVDIM / 64, S_LEN / 128), 256, 0, stream>>>(
      xb, wkb, wk_b, kbf, S_LEN, KVDIM, D_MODEL);
  gemm_xwT<true><<<dim3(KVDIM / 64, S_LEN / 128), 256, 0, stream>>>(
      xb, wvb, wv_b, vbf, S_LEN, KVDIM, D_MODEL);

  // 3) RoPE on Q and K
  {
    const int tq = S_LEN * N_HEADS * 32;
    rope_kernel<<<blocks((size_t)tq), 256, 0, stream>>>(qb, rope, N_HEADS, tq);
    const int tk = S_LEN * KV_HEADS * 32;
    rope_kernel<<<blocks((size_t)tk), 256, 0, stream>>>(kbf, rope, KV_HEADS, tk);
  }

  // 4) Sliding-window attention
  attn_kernel<<<dim3(S_LEN / 16, KV_HEADS), 256, 0, stream>>>(qb, kbf, vbf, aob);

  // 5) Output projection (f32 out with bias)
  gemm_xwT<false><<<dim3(D_MODEL / 64, S_LEN / 128), 256, 0, stream>>>(
      aob, wob, wo_b, out, S_LEN, D_MODEL, QDIM);
}
